// ScaledDotProductAttention_20280835572520
// MI455X (gfx1250) — matrix operand order repaired and checked by reading, not compiled, not run on hardware
//
#include <hip/hip_runtime.h>

// ---- problem constants ----
#define B_   16
#define LQ_  2048
#define LK_  2048
#define D_   128
#define TQ   32      // query rows per workgroup
#define KT   128     // K columns per block iteration

// LDS row strides (bf16 elements, padded, even for b32 access)
#define SQ   132
#define SK   132
#define SA   132

typedef __attribute__((ext_vector_type(16))) __bf16 v16bf;
typedef __attribute__((ext_vector_type(8)))  float  v8f;

union Frag16 { v16bf v; unsigned u[8]; };

// A-matrix 16x32 bf16 fragment (ISA 7.12.2): lane m = lane&15, half = lane>>4
// VGPR v holds K = 16*(v>>2) + 8*half + 2*(v&3) + {0,1}  -> b32 index = 8*(v>>2)+4*half+(v&3)
__device__ __forceinline__ v16bf lds_load_a16(const __bf16* base, int stride) {
  const int lane = (int)(threadIdx.x & 31u);
  const int hlf = lane >> 4, m = lane & 15;
  const unsigned* r = (const unsigned*)(base + m * stride);
  Frag16 f;
#pragma unroll
  for (int vv = 0; vv < 8; ++vv)
    f.u[vv] = r[8 * (vv >> 2) + 4 * hlf + (vv & 3)];
  return f.v;
}

// B-matrix 32x16 bf16 fragment: lane n = lane&15, half = lane>>4
// VGPR v holds K = 16*(v>>2) + 8*half + 2*(v&3) + {0,1} of column n; LDS holds B^T row-major
// (row n contiguous along K) -> b32 index = 8*(v>>2)+4*half+(v&3)
__device__ __forceinline__ v16bf lds_load_b16(const __bf16* base, int stride) {
  const int lane = (int)(threadIdx.x & 31u);
  const int hlf = lane >> 4, n = lane & 15;
  const unsigned* r = (const unsigned*)(base + n * stride);
  Frag16 f;
#pragma unroll
  for (int vv = 0; vv < 8; ++vv)
    f.u[vv] = r[8 * (vv >> 2) + 4 * hlf + (vv & 3)];
  return f.v;
}

// ---- dynamic LDS layout ----
constexpr int OFF_S   = 0;                          // fp32 S[32][2048]      = 262144 B
constexpr int OFF_KW  = OFF_S   + TQ * LK_ * 4;     // u16 keep[32][128]     =   8192 B
constexpr int OFF_Q   = OFF_KW  + TQ * (LK_ / 16) * 2;   // bf16 Q[32][SQ]   =   8448 B
constexpr int OFF_KV  = OFF_Q   + TQ * SQ * 2;      // bf16 KV[128][SK]      =  33792 B
constexpr int OFF_A   = OFF_KV  + KT * SK * 2;      // bf16 A[32][SA]        =   8448 B
constexpr int OFF_RED = OFF_A   + TQ * SA * 2;      // fp32 red[32][8]       =   1024 B
constexpr int OFF_RM  = OFF_RED + TQ * 8 * 4;       // fp32 rowmax[32]
constexpr int OFF_RS  = OFF_RM  + TQ * 4;           // fp32 rowinv[32]
constexpr int SMEM_BYTES = OFF_RS + TQ * 4;         // = 322304 B (< 320 KB WGP LDS)

__global__ __launch_bounds__(256, 1)
void sdpa_fused_kernel(const float* __restrict__ q,
                       const float* __restrict__ k,
                       const float* __restrict__ v,
                       const float* __restrict__ scaleP,
                       const float* __restrict__ mask,
                       float* __restrict__ outCtx,
                       float* __restrict__ outAttn) {
  extern __shared__ char smem[];
  float*          S      = (float*)(smem + OFF_S);
  unsigned short* keepw  = (unsigned short*)(smem + OFF_KW);
  __bf16*         Qbf    = (__bf16*)(smem + OFF_Q);
  __bf16*         KVbf   = (__bf16*)(smem + OFF_KV);
  __bf16*         Abf    = (__bf16*)(smem + OFF_A);
  float*          red    = (float*)(smem + OFF_RED);
  float*          rowm   = (float*)(smem + OFF_RM);
  float*          rowinv = (float*)(smem + OFF_RS);

  const int tid  = (int)threadIdx.x;
  const int wv   = tid >> 5;
  const int lane = tid & 31;
  const int hlf  = lane >> 4;
  const int ln   = lane & 15;

  const int b  = (int)blockIdx.x >> 6;          // batch
  const int q0 = ((int)blockIdx.x & 63) * TQ;   // query tile base

  const float scale = scaleP[0];

  // ---- stage Q tile (32x128 f32 -> bf16 LDS) ----
  {
    const int row = tid >> 3;
    const int c0  = (tid & 7) * 16;
    const float* src = q + ((size_t)b * LQ_ + q0 + row) * D_ + c0;
#pragma unroll
    for (int i = 0; i < 4; ++i) {
      float4 f = ((const float4*)src)[i];
      __bf16* dst = Qbf + row * SQ + c0 + 4 * i;
      dst[0] = (__bf16)f.x; dst[1] = (__bf16)f.y;
      dst[2] = (__bf16)f.z; dst[3] = (__bf16)f.w;
    }
  }

  const int qi = wv >> 2;   // 0..1 : which 16-row half of the query tile
  const int wq = wv & 3;    // 0..3 : column-group of this wave

  // =============== Phase 1: masked scores into LDS ===============
  for (int k0 = 0; k0 < LK_; k0 += KT) {
    if (k0 + KT < LK_)  // emits global_prefetch_b8
      __builtin_prefetch(k + ((size_t)b * LK_ + k0 + KT) * D_ + (size_t)tid * 64, 0, 0);

    // stage K block (128x128 f32 -> bf16 LDS, row-major = B^T layout for QK^T)
    {
      const int row = tid >> 1;
      const int c0  = (tid & 1) * 64;
      const float* src = k + ((size_t)b * LK_ + k0 + row) * D_ + c0;
#pragma unroll 4
      for (int i = 0; i < 16; ++i) {
        float4 f = ((const float4*)src)[i];
        __bf16* dst = KVbf + row * SK + c0 + 4 * i;
        dst[0] = (__bf16)f.x; dst[1] = (__bf16)f.y;
        dst[2] = (__bf16)f.z; dst[3] = (__bf16)f.w;
      }
    }
    // stage mask block -> keep bits (1 = keep)
    {
      const int row = tid >> 3;
      const int c0  = (tid & 7) * 16;
      const float* src = mask + ((size_t)b * LQ_ + q0 + row) * LK_ + k0 + c0;
      unsigned w16 = 0;
#pragma unroll
      for (int i = 0; i < 4; ++i) {
        float4 f = ((const float4*)src)[i];
        if (f.x < 0.5f) w16 |= 1u << (4 * i + 0);
        if (f.y < 0.5f) w16 |= 1u << (4 * i + 1);
        if (f.z < 0.5f) w16 |= 1u << (4 * i + 2);
        if (f.w < 0.5f) w16 |= 1u << (4 * i + 3);
      }
      keepw[row * (LK_ / 16) + ((k0 + c0) >> 4)] = (unsigned short)w16;
    }
    __syncthreads();

    // each wave: two 16x16 S subtiles, K=128 reduction via 4 bf16 WMMAs each
#pragma unroll
    for (int t = 0; t < 2; ++t) {
      const int kj = wq + 4 * t;
      v8f acc = {};
#pragma unroll
      for (int kk = 0; kk < 4; ++kk) {
        v16bf a  = lds_load_a16(Qbf  + (qi * 16) * SQ + kk * 32, SQ);
        v16bf bB = lds_load_b16(KVbf + (kj * 16) * SK + kk * 32, SK);
        acc = __builtin_amdgcn_wmma_f32_16x16x32_bf16(
            false, a, false, bB, (short)0, acc, false, false);
      }
      const int kcol = k0 + kj * 16 + ln;
#pragma unroll
      for (int r = 0; r < 8; ++r) {
        const int qr = qi * 16 + 8 * hlf + r;   // C/D layout: M = r + 8*half
        const bool kp = (keepw[qr * (LK_ / 16) + (kcol >> 4)] >> (kcol & 15)) & 1;
        S[qr * LK_ + kcol] = kp ? acc[r] * scale : 0.0f;
      }
    }
    __syncthreads();
  }

  // =============== Phase 1.5: row max and softmax denominator ===============
  {
    const int row = tid >> 3;
    const int seg = tid & 7;
    const float* sr = S + row * LK_ + seg * 256;
    float mx = -3.4e38f;
    for (int i = 0; i < 256; ++i) mx = fmaxf(mx, sr[i]);
    red[row * 8 + seg] = mx;
    __syncthreads();
    if (tid < TQ) {
      float m2 = red[tid * 8];
#pragma unroll
      for (int j = 1; j < 8; ++j) m2 = fmaxf(m2, red[tid * 8 + j]);
      rowm[tid] = m2;
    }
    __syncthreads();
    const float m2 = rowm[row];
    float sum = 0.0f;
    for (int i = 0; i < 256; ++i) sum += __expf(sr[i] - m2);
    red[row * 8 + seg] = sum;
    __syncthreads();
    if (tid < TQ) {
      float s2 = 0.0f;
#pragma unroll
      for (int j = 0; j < 8; ++j) s2 += red[tid * 8 + j];
      rowinv[tid] = 1.0f / s2;
    }
    __syncthreads();
  }

  // =============== Phase 2: attention out + context GEMM ===============
  v8f cacc0 = {}, cacc1 = {};
  const int nj0 = wq, nj1 = wq + 4;

  for (int k0 = 0; k0 < LK_; k0 += KT) {
    if (k0 + KT < LK_)
      __builtin_prefetch(v + ((size_t)b * LK_ + k0 + KT) * D_ + (size_t)tid * 64, 0, 0);

    // stage V block transposed (Vt[d][krow]) so B frags read rows contiguously
    {
      const int row = tid >> 1;           // krow
      const int c0  = (tid & 1) * 64;     // d
      const float* src = v + ((size_t)b * LK_ + k0 + row) * D_ + c0;
#pragma unroll 4
      for (int i = 0; i < 16; ++i) {
        float4 f = ((const float4*)src)[i];
        const int d0 = c0 + 4 * i;
        KVbf[(d0 + 0) * SK + row] = (__bf16)f.x;
        KVbf[(d0 + 1) * SK + row] = (__bf16)f.y;
        KVbf[(d0 + 2) * SK + row] = (__bf16)f.z;
        KVbf[(d0 + 3) * SK + row] = (__bf16)f.w;
      }
    }
    __syncthreads();

    // p = keep * exp(s - m) / Z  -> global attention + bf16 A tile
    {
      const int row = tid >> 3;
      const int c0  = (tid & 7) * 16;
      const float mrow = rowm[row];
      const float rinv = rowinv[row];
      float pbuf[16];
#pragma unroll
      for (int i = 0; i < 16; ++i) {
        const int gcol = k0 + c0 + i;
        const float sv = S[row * LK_ + gcol];
        const bool kp = (keepw[row * (LK_ / 16) + (gcol >> 4)] >> (gcol & 15)) & 1;
        const float p = kp ? __expf(sv - mrow) * rinv : 0.0f;
        pbuf[i] = p;
        Abf[row * SA + c0 + i] = (__bf16)p;
      }
      float4* dst = (float4*)(outAttn + ((size_t)b * LQ_ + q0 + row) * LK_ + k0 + c0);
#pragma unroll
      for (int i = 0; i < 4; ++i)
        dst[i] = make_float4(pbuf[4 * i], pbuf[4 * i + 1], pbuf[4 * i + 2], pbuf[4 * i + 3]);
    }
    __syncthreads();

    // C[32x128] += A[32x128] * V[128x128]
#pragma unroll
    for (int kk = 0; kk < 4; ++kk) {
      v16bf a  = lds_load_a16(Abf  + (qi * 16) * SA + kk * 32, SA);
      v16bf b0 = lds_load_b16(KVbf + (nj0 * 16) * SK + kk * 32, SK);
      v16bf b1 = lds_load_b16(KVbf + (nj1 * 16) * SK + kk * 32, SK);
      cacc0 = __builtin_amdgcn_wmma_f32_16x16x32_bf16(
          false, a, false, b0, (short)0, cacc0, false, false);
      cacc1 = __builtin_amdgcn_wmma_f32_16x16x32_bf16(
          false, a, false, b1, (short)0, cacc1, false, false);
    }
    __syncthreads();
  }

  // ---- write context (C/D layout: M = r + 8*half, N = lane&15) ----
#pragma unroll
  for (int r = 0; r < 8; ++r) {
    const int qr = q0 + qi * 16 + 8 * hlf + r;
    outCtx[((size_t)b * LQ_ + qr) * D_ + nj0 * 16 + ln] = cacc0[r];
    outCtx[((size_t)b * LQ_ + qr) * D_ + nj1 * 16 + ln] = cacc1[r];
  }
}

extern "C" void kernel_launch(void* const* d_in, const int* in_sizes, int n_in,
                              void* d_out, int out_size, void* d_ws, size_t ws_size,
                              hipStream_t stream) {
  (void)in_sizes; (void)n_in; (void)out_size; (void)d_ws; (void)ws_size;
  const float* q     = (const float*)d_in[0];
  const float* k     = (const float*)d_in[1];
  const float* v     = (const float*)d_in[2];
  const float* scale = (const float*)d_in[3];
  const float* mask  = (const float*)d_in[4];
  float* outCtx  = (float*)d_out;
  float* outAttn = outCtx + (size_t)B_ * LQ_ * D_;   // DV == D_ == 128

  dim3 grid(B_ * (LQ_ / TQ));   // 16 * 64 = 1024 workgroups
  dim3 block(256);              // 8 wave32
  sdpa_fused_kernel<<<grid, block, SMEM_BYTES, stream>>>(q, k, v, scale, mask,
                                                         outCtx, outAttn);
}
